// Encoder_10900626998086
// MI455X (gfx1250) — compile-verified, matrix operand order checked by reading, not run on hardware
//
#include <hip/hip_runtime.h>
#include <cstdint>
#include <cstddef>

// ---------------- model constants ----------------
#define Lc 12
#define Hc 12
#define Dc 768
#define DHc 64
#define FFc 3072
#define Bc 8
#define Sc 512
#define Mc (Bc * Sc)          // 4096 rows
#define SCALEc 0.125f         // 64^-0.5

// ---------------- WMMA types ----------------
typedef __attribute__((ext_vector_type(16))) __bf16 v16bf;
typedef __attribute__((ext_vector_type(8)))  float  v8f;

union BF16x16 { uint4 u[2]; v16bf v; };

static __device__ inline v8f wmma_bf16(const BF16x16& a, const BF16x16& b, v8f c) {
  return __builtin_amdgcn_wmma_f32_16x16x32_bf16(false, a.v, false, b.v, (short)0, c, false, false);
}

static __device__ inline uint16_t f2bf(float f) {
  uint32_t u = __builtin_bit_cast(uint32_t, f);
  uint32_t r = (u + 0x7fffu + ((u >> 16) & 1u)) >> 16;
  return (uint16_t)r;
}

// ---- CDNA5 async global->LDS copy (16 bytes per lane), tracked by ASYNCcnt ----
static __device__ inline void async_copy_b128(const void* gptr, void* ldsptr) {
  uint64_t ga = (uint64_t)(uintptr_t)gptr;
  uint32_t la = (uint32_t)(uintptr_t)ldsptr;   // LDS offset = low 32 bits of generic ptr
  asm volatile("global_load_async_to_lds_b128 %0, %1, off"
               :: "v"(la), "v"(ga) : "memory");
}
static __device__ inline void wait_async0() {
  asm volatile("s_wait_asynccnt 0" ::: "memory");
}

// ---------------- embeddings + key-pad mask ----------------
__global__ __launch_bounds__(256) void embed_kernel(
    const int* __restrict__ inputs, const int* __restrict__ segments,
    const float* __restrict__ we, const float* __restrict__ pe,
    const float* __restrict__ te, float* __restrict__ xf,
    uint16_t* __restrict__ xb, float* __restrict__ mask)
{
  int bs = blockIdx.x;                 // 0..4095
  int s = bs & (Sc - 1);
  int id = inputs[bs];
  int seg = segments[bs];
  int pos = (id == 0) ? 0 : (s + 1);
  const float* wr = we + (size_t)id  * Dc;
  const float* pr = pe + (size_t)pos * Dc;
  const float* tr = te + (size_t)seg * Dc;
  for (int d = threadIdx.x; d < Dc; d += 256) {
    float v = wr[d] + pr[d] + tr[d];
    xf[(size_t)bs * Dc + d] = v;
    xb[(size_t)bs * Dc + d] = f2bf(v);
  }
  if (threadIdx.x == 0) mask[bs] = (id == 0) ? -1e9f : 0.0f;
}

// ---------------- fp32 [K,N] -> bf16 transposed [N,K] ----------------
__global__ void tconv_kernel(const float* __restrict__ W, uint16_t* __restrict__ Wt,
                             int K, int N)
{
  __shared__ float tile[32][33];
  int nb = blockIdx.x * 32, kb = blockIdx.y * 32;
  int tx = threadIdx.x, ty = threadIdx.y;           // (32, 8)
  for (int i = 0; i < 32; i += 8)
    tile[ty + i][tx] = W[(size_t)(kb + ty + i) * N + nb + tx];
  __syncthreads();
  for (int i = 0; i < 32; i += 8)
    Wt[(size_t)(nb + ty + i) * K + kb + tx] = f2bf(tile[tx][ty + i]);
}

// ---------------- tiled WMMA GEMM: C[M,N] = A[M,K](bf16) * Wt[N,K](bf16) + bias ----
// Double-buffered async global->LDS staging (CDNA5 ASYNCcnt path).
// mode 0: fp32 out ; 1: bf16 out ; 2: gelu->bf16 out ; 3: bf16 per-head transposed (V)
__global__ __launch_bounds__(256) void gemm_kernel(
    const uint16_t* __restrict__ A, const uint16_t* __restrict__ Bt,
    const float* __restrict__ bias, float* __restrict__ outf,
    uint16_t* __restrict__ outb, int M, int N, int K, int mode)
{
  __shared__ __align__(16) uint16_t As[2][128 * 40];
  __shared__ __align__(16) uint16_t Bs[2][128 * 40];
  int tid = threadIdx.x, lane = tid & 31, w = tid >> 5;
  int wm = w >> 1, wn = w & 1;
  int m0 = blockIdx.y * 128, n0 = blockIdx.x * 128;
  int half = lane >> 4, lr = lane & 15;

  v8f acc[2][4];
  v8f zero = {0.f, 0.f, 0.f, 0.f, 0.f, 0.f, 0.f, 0.f};
  for (int i = 0; i < 2; i++)
    for (int j = 0; j < 4; j++) acc[i][j] = zero;

  int lrow = tid >> 1;
  int lch  = (tid & 1) * 16;
  const uint16_t* ga = A  + (size_t)(m0 + lrow) * K + lch;
  const uint16_t* gb = Bt + (size_t)(n0 + lrow) * K + lch;

  // issue async staging of one 128x32 A tile + 128x32 B tile into buffer `buf`
  auto issue = [&](int buf, int kk) {
    uint16_t* la = &As[buf][lrow * 40 + lch];
    uint16_t* lb = &Bs[buf][lrow * 40 + lch];
    async_copy_b128(ga + kk,     la);
    async_copy_b128(ga + kk + 8, la + 8);
    async_copy_b128(gb + kk,     lb);
    async_copy_b128(gb + kk + 8, lb + 8);
  };

  int nk = K >> 5;
  issue(0, 0);
  wait_async0();
  __syncthreads();

  for (int t = 0; t < nk; t++) {
    int cur = t & 1;
    if (t + 1 < nk) issue(cur ^ 1, (t + 1) * 32);   // prefetch next tile (other buffer)

    BF16x16 a[2], bb[4];
    #pragma unroll
    for (int i = 0; i < 2; i++) {
      const uint4* p = (const uint4*)(&As[cur][(wm * 32 + i * 16 + lr) * 40]);
      a[i].u[0] = p[half]; a[i].u[1] = p[2 + half];
    }
    #pragma unroll
    for (int j = 0; j < 4; j++) {
      const uint4* p = (const uint4*)(&Bs[cur][(wn * 64 + j * 16 + lr) * 40]);
      bb[j].u[0] = p[half]; bb[j].u[1] = p[2 + half];
    }
    #pragma unroll
    for (int i = 0; i < 2; i++)
      #pragma unroll
      for (int j = 0; j < 4; j++)
        acc[i][j] = wmma_bf16(a[i], bb[j], acc[i][j]);

    wait_async0();      // next-tile async loads landed (overlapped with WMMAs above)
    __syncthreads();    // all waves' LDS writes visible / reads of `cur` finished
  }

  // epilogue
  #pragma unroll
  for (int i = 0; i < 2; i++) {
    #pragma unroll
    for (int j = 0; j < 4; j++) {
      int n = n0 + wn * 64 + j * 16 + lr;
      float bv = bias[n];
      #pragma unroll
      for (int r = 0; r < 8; r++) {
        int m = m0 + wm * 32 + i * 16 + r + 8 * half;
        float v = acc[i][j][r] + bv;
        if (mode == 0) {
          outf[(size_t)m * N + n] = v;
        } else if (mode == 1) {
          outb[(size_t)m * N + n] = f2bf(v);
        } else if (mode == 2) {
          v = 0.5f * v * (1.0f + erff(v * 0.70710678f));
          outb[(size_t)m * N + n] = f2bf(v);
        } else { // V -> Vt[b][h][dh][s]
          int hh = n / DHc, dh = n % DHc;
          int b = m >> 9, s = m & (Sc - 1);
          outb[(((size_t)b * Hc + hh) * DHc + dh) * Sc + s] = f2bf(v);
        }
      }
    }
  }
}

// ---------------- fused flash-style attention ----------------
// grid: (S/64, H, B), block 128 (4 waves). Each wave: 16 query rows.
__global__ __launch_bounds__(128) void attn_kernel(
    const uint16_t* __restrict__ Q, const uint16_t* __restrict__ Kb,
    const uint16_t* __restrict__ Vt, const float* __restrict__ mask,
    uint16_t* __restrict__ ctx)
{
  __shared__ __align__(16) uint16_t plds[4][16 * 32];
  int lane = threadIdx.x & 31, w = threadIdx.x >> 5;
  int h = blockIdx.y, b = blockIdx.z;
  int qb = blockIdx.x * 64 + w * 16;
  int half = lane >> 4, lr = lane & 15;

  BF16x16 a_lo, a_hi;
  { // Q tile, lives in registers for whole kernel
    const uint4* p = (const uint4*)(Q + ((size_t)(b * Sc + qb + lr) * Dc + h * DHc));
    a_lo.u[0] = p[half];     a_lo.u[1] = p[2 + half];
    a_hi.u[0] = p[4 + half]; a_hi.u[1] = p[6 + half];
  }

  float Mrow[8], Srow[8];
  v8f zero = {0.f, 0.f, 0.f, 0.f, 0.f, 0.f, 0.f, 0.f};
  v8f acc[4];
  #pragma unroll
  for (int t = 0; t < 4; t++) acc[t] = zero;
  #pragma unroll
  for (int r = 0; r < 8; r++) { Mrow[r] = -3.0e38f; Srow[r] = 0.0f; }

  uint16_t* pl = plds[w];
  const float* mrow = mask + (size_t)b * Sc;

  for (int kb = 0; kb < Sc; kb += 32) {
    v8f s0 = zero, s1 = zero;
    {
      BF16x16 b0lo, b0hi;
      const uint4* p = (const uint4*)(Kb + ((size_t)(b * Sc + kb + lr) * Dc + h * DHc));
      b0lo.u[0] = p[half];     b0lo.u[1] = p[2 + half];
      b0hi.u[0] = p[4 + half]; b0hi.u[1] = p[6 + half];
      s0 = wmma_bf16(a_lo, b0lo, s0);
      s0 = wmma_bf16(a_hi, b0hi, s0);
    }
    {
      BF16x16 b1lo, b1hi;
      const uint4* p = (const uint4*)(Kb + ((size_t)(b * Sc + kb + 16 + lr) * Dc + h * DHc));
      b1lo.u[0] = p[half];     b1lo.u[1] = p[2 + half];
      b1hi.u[0] = p[4 + half]; b1hi.u[1] = p[6 + half];
      s1 = wmma_bf16(a_lo, b1lo, s1);
      s1 = wmma_bf16(a_hi, b1hi, s1);
    }
    float mv0 = mrow[kb + lr];
    float mv1 = mrow[kb + 16 + lr];

    #pragma unroll
    for (int r = 0; r < 8; r++) {
      float x0 = s0[r] * SCALEc + mv0;
      float x1 = s1[r] * SCALEc + mv1;
      float mx = fmaxf(x0, x1);
      #pragma unroll
      for (int off = 1; off < 16; off <<= 1)
        mx = fmaxf(mx, __shfl_xor(mx, off, 32));
      float newM = fmaxf(Mrow[r], mx);
      float corr = expf(Mrow[r] - newM);
      float p0 = expf(x0 - newM);
      float p1 = expf(x1 - newM);
      float ps = p0 + p1;
      #pragma unroll
      for (int off = 1; off < 16; off <<= 1)
        ps += __shfl_xor(ps, off, 32);
      Srow[r] = Srow[r] * corr + ps;
      Mrow[r] = newM;
      #pragma unroll
      for (int t = 0; t < 4; t++) acc[t][r] *= corr;
      int rowl = r + 8 * half;
      pl[rowl * 32 + lr]      = f2bf(p0);
      pl[rowl * 32 + 16 + lr] = f2bf(p1);
    }

    BF16x16 pa; // probs A-tile from LDS (same-wave DS ops are in order)
    {
      const uint4* p = (const uint4*)(pl + lr * 32);
      pa.u[0] = p[half]; pa.u[1] = p[2 + half];
    }
    #pragma unroll
    for (int t = 0; t < 4; t++) {
      BF16x16 bv;
      const uint4* p = (const uint4*)(Vt + (((size_t)(b * Hc + h) * DHc + t * 16 + lr) * Sc + kb));
      bv.u[0] = p[half]; bv.u[1] = p[2 + half];
      acc[t] = wmma_bf16(pa, bv, acc[t]);
    }
  }

  #pragma unroll
  for (int t = 0; t < 4; t++) {
    #pragma unroll
    for (int r = 0; r < 8; r++) {
      int qrow = qb + r + 8 * half;
      float v = acc[t][r] / Srow[r];
      ctx[(size_t)(b * Sc + qrow) * Dc + h * DHc + t * 16 + lr] = f2bf(v);
    }
  }
}

// ---------------- residual add + LayerNorm (row per block) ----------------
__global__ __launch_bounds__(256) void ln_kernel(
    const float* __restrict__ xin, const float* __restrict__ res,
    const float* __restrict__ g, const float* __restrict__ bta,
    float* __restrict__ xout, uint16_t* __restrict__ xbf)
{
  __shared__ float sd[256];
  int row = blockIdx.x, t = threadIdx.x;
  const float* xr = xin + (size_t)row * Dc;
  const float* rr = res + (size_t)row * Dc;
  float v[3];
  float s = 0.f;
  #pragma unroll
  for (int i = 0; i < 3; i++) { v[i] = xr[t + i * 256] + rr[t + i * 256]; s += v[i]; }
  sd[t] = s; __syncthreads();
  for (int o = 128; o > 0; o >>= 1) { if (t < o) sd[t] += sd[t + o]; __syncthreads(); }
  float mean = sd[0] * (1.0f / Dc);
  __syncthreads();
  s = 0.f;
  #pragma unroll
  for (int i = 0; i < 3; i++) { float d = v[i] - mean; s += d * d; }
  sd[t] = s; __syncthreads();
  for (int o = 128; o > 0; o >>= 1) { if (t < o) sd[t] += sd[t + o]; __syncthreads(); }
  float rstd = rsqrtf(sd[0] * (1.0f / Dc) + 1e-12f);
  #pragma unroll
  for (int i = 0; i < 3; i++) {
    int c = t + i * 256;
    float o = (v[i] - mean) * rstd * g[c] + bta[c];
    xout[(size_t)row * Dc + c] = o;
    xbf[(size_t)row * Dc + c] = f2bf(o);
  }
}

// ---------------- host ----------------
static inline char* ws_alloc(char*& p, size_t bytes) {
  char* r = p;
  p += (bytes + 255) & ~(size_t)255;
  return r;
}

extern "C" void kernel_launch(void* const* d_in, const int* in_sizes, int n_in,
                              void* d_out, int out_size, void* d_ws, size_t ws_size,
                              hipStream_t stream) {
  const int*   inputs   = (const int*)d_in[0];
  const int*   segments = (const int*)d_in[1];
  const float* we  = (const float*)d_in[2];
  const float* pe  = (const float*)d_in[3];
  const float* te  = (const float*)d_in[4];
  const float* Wq  = (const float*)d_in[5];
  const float* bq  = (const float*)d_in[6];
  const float* Wk  = (const float*)d_in[7];
  const float* bk  = (const float*)d_in[8];
  const float* Wv  = (const float*)d_in[9];
  const float* bvp = (const float*)d_in[10];
  const float* Wo  = (const float*)d_in[11];
  const float* bo  = (const float*)d_in[12];
  const float* W1  = (const float*)d_in[13];
  const float* bf1 = (const float*)d_in[14];
  const float* W2  = (const float*)d_in[15];
  const float* bf2 = (const float*)d_in[16];
  const float* g1  = (const float*)d_in[17];
  const float* b1  = (const float*)d_in[18];
  const float* g2  = (const float*)d_in[19];
  const float* b2  = (const float*)d_in[20];

  char* p = (char*)d_ws;
  float*    xf   = (float*)   ws_alloc(p, (size_t)Mc * Dc * 4);
  float*    tmp  = (float*)   ws_alloc(p, (size_t)Mc * Dc * 4);
  float*    mask = (float*)   ws_alloc(p, (size_t)Mc * 4);
  uint16_t* xb   = (uint16_t*)ws_alloc(p, (size_t)Mc * Dc * 2);
  uint16_t* qb   = (uint16_t*)ws_alloc(p, (size_t)Mc * Dc * 2);
  uint16_t* kbuf = (uint16_t*)ws_alloc(p, (size_t)Mc * Dc * 2);
  uint16_t* vt   = (uint16_t*)ws_alloc(p, (size_t)Mc * Dc * 2);
  uint16_t* ctx  = (uint16_t*)ws_alloc(p, (size_t)Mc * Dc * 2);
  uint16_t* hb   = (uint16_t*)ws_alloc(p, (size_t)Mc * FFc * 2);
  uint16_t* wqt  = (uint16_t*)ws_alloc(p, (size_t)Dc * Dc * 2);
  uint16_t* wkt  = (uint16_t*)ws_alloc(p, (size_t)Dc * Dc * 2);
  uint16_t* wvt  = (uint16_t*)ws_alloc(p, (size_t)Dc * Dc * 2);
  uint16_t* wot  = (uint16_t*)ws_alloc(p, (size_t)Dc * Dc * 2);
  uint16_t* w1t  = (uint16_t*)ws_alloc(p, (size_t)Dc * FFc * 2);
  uint16_t* w2t  = (uint16_t*)ws_alloc(p, (size_t)FFc * Dc * 2);

  embed_kernel<<<Mc, 256, 0, stream>>>(inputs, segments, we, pe, te, xf, xb, mask);

  dim3 tcb(32, 8);
  dim3 gemmB(256);

  for (int l = 0; l < Lc; l++) {
    const float* wq_l = Wq + (size_t)l * Dc * Dc;
    const float* wk_l = Wk + (size_t)l * Dc * Dc;
    const float* wv_l = Wv + (size_t)l * Dc * Dc;
    const float* wo_l = Wo + (size_t)l * Dc * Dc;
    const float* w1_l = W1 + (size_t)l * Dc * FFc;
    const float* w2_l = W2 + (size_t)l * FFc * Dc;

    tconv_kernel<<<dim3(Dc / 32, Dc / 32), tcb, 0, stream>>>(wq_l, wqt, Dc, Dc);
    tconv_kernel<<<dim3(Dc / 32, Dc / 32), tcb, 0, stream>>>(wk_l, wkt, Dc, Dc);
    tconv_kernel<<<dim3(Dc / 32, Dc / 32), tcb, 0, stream>>>(wv_l, wvt, Dc, Dc);
    tconv_kernel<<<dim3(Dc / 32, Dc / 32), tcb, 0, stream>>>(wo_l, wot, Dc, Dc);
    tconv_kernel<<<dim3(FFc / 32, Dc / 32), tcb, 0, stream>>>(w1_l, w1t, Dc, FFc);
    tconv_kernel<<<dim3(Dc / 32, FFc / 32), tcb, 0, stream>>>(w2_l, w2t, FFc, Dc);

    // Q, K, V projections
    gemm_kernel<<<dim3(Dc / 128, Mc / 128), gemmB, 0, stream>>>(
        xb, wqt, bq + l * Dc, nullptr, qb, Mc, Dc, Dc, 1);
    gemm_kernel<<<dim3(Dc / 128, Mc / 128), gemmB, 0, stream>>>(
        xb, wkt, bk + l * Dc, nullptr, kbuf, Mc, Dc, Dc, 1);
    gemm_kernel<<<dim3(Dc / 128, Mc / 128), gemmB, 0, stream>>>(
        xb, wvt, bvp + l * Dc, nullptr, vt, Mc, Dc, Dc, 3);

    attn_kernel<<<dim3(Sc / 64, Hc, Bc), dim3(128), 0, stream>>>(qb, kbuf, vt, mask, ctx);

    gemm_kernel<<<dim3(Dc / 128, Mc / 128), gemmB, 0, stream>>>(
        ctx, wot, bo + l * Dc, tmp, nullptr, Mc, Dc, Dc, 0);
    ln_kernel<<<Mc, 256, 0, stream>>>(xf, tmp, g1 + l * Dc, b1 + l * Dc, xf, xb);

    gemm_kernel<<<dim3(FFc / 128, Mc / 128), gemmB, 0, stream>>>(
        xb, w1t, bf1 + l * FFc, nullptr, hb, Mc, FFc, Dc, 2);
    gemm_kernel<<<dim3(Dc / 128, Mc / 128), gemmB, 0, stream>>>(
        hb, w2t, bf2 + l * Dc, tmp, nullptr, Mc, Dc, FFc, 0);

    float* xo = (l == Lc - 1) ? (float*)d_out : xf;
    ln_kernel<<<Mc, 256, 0, stream>>>(xf, tmp, g2 + l * Dc, b2 + l * Dc, xo, xb);
  }
}
